// EvoEncoder_30760555774048
// MI455X (gfx1250) — matrix operand order repaired and checked by reading, not compiled, not run on hardware
//
#include <hip/hip_runtime.h>
#include <hip/hip_bf16.h>

// ---------------------------------------------------------------------------
// Types for CDNA5 WMMA (wave32): V_WMMA_F32_16X16X32_BF16
// ---------------------------------------------------------------------------
typedef __attribute__((ext_vector_type(16))) __bf16 v16bf;
typedef __attribute__((ext_vector_type(8)))  float  v8f;

// Native f32 -> bf16 conversion: lowers to gfx1250 hardware cvt (packed where
// possible) instead of a 3-VALU-op software round-to-nearest-even sequence.
__device__ __forceinline__ __bf16 f2bf(float f) { return (__bf16)f; }

// ---------------------------------------------------------------------------
// Generic batched GEMM on the WMMA pipe, 32x32 tile per wave (2x2 WMMA block).
//   C[b1,b2] = act( alpha * op(A[b1,b2]) @ op(B[b1,b2]) + bias )
//   !TA: A is [M,K] lda ; TA: A is [K,M] lda
//   !TB: B is [K,N] ldb ; TB: B is [N,K] ldb
//   Requires M%32==0, K%32==0. FULLN=true requires N%32==0 (host-dispatched);
//   FULLN=false guards N (used only for tiny N=20/8/4 GEMMs).
// ACT: 0 = none, 1 = relu, 2 = sigmoid
// ---------------------------------------------------------------------------
template<int ACT, bool TA, bool TB, bool FULLN>
__global__ __launch_bounds__(32)
void gemm_k(const float* __restrict__ A, const float* __restrict__ B,
            const float* __restrict__ bias, float* __restrict__ C,
            int M, int N, int K, int lda, int ldb, int ldc,
            long long sA1, long long sA2, long long sB1, long long sB2,
            long long sC1, long long sC2, float alpha)
{
  const int b1 = blockIdx.z, b2 = blockIdx.y;
  A += (long long)b1 * sA1 + (long long)b2 * sA2;
  B += (long long)b1 * sB1 + (long long)b2 * sB2;
  C += (long long)b1 * sC1 + (long long)b2 * sC2;

  const int tilesN = (N + 31) >> 5;
  const int tm = blockIdx.x / tilesN;
  const int tn = blockIdx.x % tilesN;
  const int lane = threadIdx.x;

  // A fragment: row m = lane&15 ; K groups 0..7 & 16..23 (lanes<16), 8..15 & 24..31 (lanes>=16)
  const int mrow = lane & 15;
  const int kgrp = (lane >> 4) * 8;
  const int m0 = tm * 32 + mrow;
  const int m1 = m0 + 16;
  // B fragment: col n = lane&15 ; K order same as A fragment
  const int n0 = tn * 32;

  v8f acc00 = {}, acc01 = {}, acc10 = {}, acc11 = {};
  for (int k0 = 0; k0 < K; k0 += 32) {
    if (k0 + 32 < K) {   // gfx1250 global_prefetch_b8 of the next K panel
      const float* pa = TA ? &A[(long long)(k0 + 32 + kgrp) * lda + m0]
                           : &A[(long long)m0 * lda + (k0 + 32 + kgrp)];
      __builtin_prefetch(pa, 0, 1);
      const float* pb = TB ? &B[(long long)n0 * ldb + (k0 + 32 + lane)]
                           : &B[(long long)(k0 + 32 + lane) * ldb + n0];
      __builtin_prefetch(pb, 0, 1);
    }
    v16bf fa0, fa1, fb0, fb1;
#pragma unroll
    for (int e = 0; e < 8; ++e) {
      const int kk  = k0 + kgrp + e;
      const int kk2 = kk + 16;
      fa0[e]     = f2bf(TA ? A[(long long)kk  * lda + m0] : A[(long long)m0 * lda + kk ]);
      fa0[e + 8] = f2bf(TA ? A[(long long)kk2 * lda + m0] : A[(long long)m0 * lda + kk2]);
      fa1[e]     = f2bf(TA ? A[(long long)kk  * lda + m1] : A[(long long)m1 * lda + kk ]);
      fa1[e + 8] = f2bf(TA ? A[(long long)kk2 * lda + m1] : A[(long long)m1 * lda + kk2]);
    }
#pragma unroll
    for (int e = 0; e < 16; ++e) {
      const int kb = k0 + kgrp + (e < 8 ? e : e + 8);
      const int na = n0 + mrow, nb = n0 + 16 + mrow;
      float v0, v1;
      if (FULLN) {
        v0 = TB ? B[(long long)na * ldb + kb] : B[(long long)kb * ldb + na];
        v1 = TB ? B[(long long)nb * ldb + kb] : B[(long long)kb * ldb + nb];
      } else {
        v0 = (na < N) ? (TB ? B[(long long)na * ldb + kb] : B[(long long)kb * ldb + na]) : 0.f;
        v1 = (nb < N) ? (TB ? B[(long long)nb * ldb + kb] : B[(long long)kb * ldb + nb]) : 0.f;
      }
      fb0[e] = f2bf(v0);
      fb1[e] = f2bf(v1);
    }
    acc00 = __builtin_amdgcn_wmma_f32_16x16x32_bf16(false, fa0, false, fb0, (short)0, acc00, false, false);
    acc01 = __builtin_amdgcn_wmma_f32_16x16x32_bf16(false, fa0, false, fb1, (short)0, acc01, false, false);
    acc10 = __builtin_amdgcn_wmma_f32_16x16x32_bf16(false, fa1, false, fb0, (short)0, acc10, false, false);
    acc11 = __builtin_amdgcn_wmma_f32_16x16x32_bf16(false, fa1, false, fb1, (short)0, acc11, false, false);
  }

  // C/D layout per 16x16 subtile: col = lane&15 ; row = +(lane>>4)*8 + r
  const int csub = lane & 15;
  const int rsub = (lane >> 4) * 8;
#pragma unroll
  for (int i = 0; i < 2; ++i) {
#pragma unroll
    for (int j = 0; j < 2; ++j) {
      const v8f& acc = (i == 0) ? (j == 0 ? acc00 : acc01) : (j == 0 ? acc10 : acc11);
      const int nc = n0 + j * 16 + csub;
      if (FULLN || nc < N) {
        const float bb = bias ? bias[nc] : 0.f;
        const int rbase = tm * 32 + i * 16 + rsub;
#pragma unroll
        for (int r = 0; r < 8; ++r) {
          float v = acc[r] * alpha + bb;
          if (ACT == 1) v = fmaxf(v, 0.f);
          else if (ACT == 2) v = 1.f / (1.f + __expf(-v));
          C[(long long)(rbase + r) * ldc + nc] = v;
        }
      }
    }
  }
}

// ---------------------------------------------------------------------------
// LayerNorm over last dim C (one block of 128 threads per row)
// ---------------------------------------------------------------------------
__global__ __launch_bounds__(128)
void layernorm_k(const float* __restrict__ x, const float* __restrict__ g,
                 const float* __restrict__ b, float* __restrict__ y, int C)
{
  __shared__ float r1[128], r2[128];
  const long long row = blockIdx.x;
  const float* xr = x + row * C;
  float* yr = y + row * C;
  const int tid = threadIdx.x;
  float s = 0.f, s2 = 0.f;
  for (int c = tid; c < C; c += 128) { float v = xr[c]; s += v; s2 += v * v; }
  r1[tid] = s; r2[tid] = s2; __syncthreads();
  for (int st = 64; st > 0; st >>= 1) {
    if (tid < st) { r1[tid] += r1[tid + st]; r2[tid] += r2[tid + st]; }
    __syncthreads();
  }
  const float mu  = r1[0] / C;
  const float var = r2[0] / C - mu * mu;
  const float inv = rsqrtf(var + 1e-5f);
  for (int c = tid; c < C; c += 128)
    yr[c] = (xr[c] - mu) * inv * g[c] + b[c];
}

// ---------------------------------------------------------------------------
// Row softmax with optional additive bias broadcast over the leading batch:
//   x rows ordered [b1, b2, m, :], bias indexed [b2, m, :]
// ---------------------------------------------------------------------------
__global__ __launch_bounds__(128)
void softmax_k(float* __restrict__ x, const float* __restrict__ bias,
               int B2, int M, int ncols)
{
  __shared__ float red[128];
  const long long row = blockIdx.x;
  float* xr = x + row * (long long)ncols;
  const float* br = nullptr;
  if (bias) {
    int m  = (int)(row % M);
    int b2 = (int)((row / M) % B2);
    br = bias + ((long long)b2 * M + m) * (long long)ncols;
  }
  const int tid = threadIdx.x;
  float mx = -3.4e38f;
  for (int c = tid; c < ncols; c += 128) {
    float v = xr[c] + (br ? br[c] : 0.f);
    mx = fmaxf(mx, v);
  }
  red[tid] = mx; __syncthreads();
  for (int s = 64; s > 0; s >>= 1) {
    if (tid < s) red[tid] = fmaxf(red[tid], red[tid + s]);
    __syncthreads();
  }
  mx = red[0]; __syncthreads();
  float sum = 0.f;
  for (int c = tid; c < ncols; c += 128) {
    float v = __expf(xr[c] + (br ? br[c] : 0.f) - mx);
    xr[c] = v; sum += v;
  }
  red[tid] = sum; __syncthreads();
  for (int s = 64; s > 0; s >>= 1) {
    if (tid < s) red[tid] += red[tid + s];
    __syncthreads();
  }
  const float inv = 1.f / red[0];
  for (int c = tid; c < ncols; c += 128) xr[c] *= inv;
}

// ---------------------------------------------------------------------------
// Elementwise helpers
// ---------------------------------------------------------------------------
__global__ void accum_k(float* d, const float* s, long long n) {
  long long i = blockIdx.x * 256LL + threadIdx.x; if (i < n) d[i] += s[i];
}
__global__ void accum_scaled_k(float* d, const float* s, float a, long long n) {
  long long i = blockIdx.x * 256LL + threadIdx.x; if (i < n) d[i] += a * s[i];
}
__global__ void mul_k(float* o, const float* a, const float* b, long long n) {
  long long i = blockIdx.x * 256LL + threadIdx.x; if (i < n) o[i] = a[i] * b[i];
}
__global__ void add2_k(float* o, const float* a, const float* b, long long n) {
  long long i = blockIdx.x * 256LL + threadIdx.x; if (i < n) o[i] = a[i] + b[i];
}
// out[j,i,c] = in[i,j,c]
__global__ void permute01_k(const float* __restrict__ in, float* __restrict__ out,
                            int N1, int N2, int C) {
  long long idx = blockIdx.x * 256LL + threadIdx.x;
  long long n = (long long)N1 * N2 * C;
  if (idx >= n) return;
  int c = (int)(idx % C);
  long long t = idx / C;
  int j = (int)(t % N2);
  int i = (int)(t / N2);
  out[((long long)j * N1 + i) * C + c] = in[idx];
}
// out[C,R] = in[R,C]^T, LDS-tiled
__global__ __launch_bounds__(256)
void transpose_k(const float* __restrict__ in, float* __restrict__ out, int R, int C) {
  __shared__ float tile[32][33];
  const int cb = blockIdx.x * 32, rb = blockIdx.y * 32;
  for (int i = threadIdx.y; i < 32; i += 8) {
    int r = rb + i, c = cb + threadIdx.x;
    tile[i][threadIdx.x] = (r < R && c < C) ? in[(long long)r * C + c] : 0.f;
  }
  __syncthreads();
  for (int i = threadIdx.y; i < 32; i += 8) {
    int c = cb + i, r = rb + threadIdx.x;
    if (c < C && r < R) out[(long long)c * R + r] = tile[threadIdx.x][i];
  }
}

// ---------------------------------------------------------------------------
// Host-side GEMM dispatcher
// ---------------------------------------------------------------------------
static void gemm_launch(hipStream_t st, const float* A, const float* B,
                        const float* bias, float* C,
                        int M, int N, int K, int lda, int ldb, int ldc,
                        long long sA1, long long sA2, long long sB1, long long sB2,
                        long long sC1, long long sC2, int NB1, int NB2,
                        float alpha, int act, int ta, int tb)
{
  const int tilesM = (M + 31) / 32, tilesN = (N + 31) / 32;
  dim3 grid(tilesM * tilesN, NB2, NB1), block(32);
  const bool fn = (N % 32) == 0;
#define GC(a, t1, t2)                                                          \
  do {                                                                         \
    if (fn) gemm_k<a, t1, t2, true ><<<grid, block, 0, st>>>(                  \
        A, B, bias, C, M, N, K, lda, ldb, ldc, sA1, sA2, sB1, sB2, sC1, sC2, alpha); \
    else    gemm_k<a, t1, t2, false><<<grid, block, 0, st>>>(                  \
        A, B, bias, C, M, N, K, lda, ldb, ldc, sA1, sA2, sB1, sB2, sC1, sC2, alpha); \
  } while (0)
  switch (act * 4 + ta * 2 + tb) {
    case 0:  GC(0, false, false); break;
    case 1:  GC(0, false, true ); break;
    case 2:  GC(0, true , false); break;
    case 3:  GC(0, true , true ); break;
    case 4:  GC(1, false, false); break;
    case 5:  GC(1, false, true ); break;
    case 6:  GC(1, true , false); break;
    case 7:  GC(1, true , true ); break;
    case 8:  GC(2, false, false); break;
    case 9:  GC(2, false, true ); break;
    case 10: GC(2, true , false); break;
    default: GC(2, true , true ); break;
  }
#undef GC
}

// ---------------------------------------------------------------------------
// Orchestration
// ---------------------------------------------------------------------------
extern "C" void kernel_launch(void* const* d_in, const int* in_sizes, int n_in,
                              void* d_out, int out_size, void* d_ws, size_t ws_size,
                              hipStream_t stream) {
  (void)in_sizes; (void)n_in; (void)out_size; (void)ws_size;
  const int S = 128, NR = 256, CM = 384, CZ = 128, HM = 8, HP = 4, NT = 4;
  const int HCM = 256;   // H_MSA * C_MSA
  const int HCP = 128;   // H_PAIR * C_PAIR
  const long long SN = (long long)S * NR;   // 32768
  const long long NN = (long long)NR * NR;  // 65536
  const float qscale = 0.1767766953f;       // 32^-0.5

  const float* msa  = (const float*)d_in[0];
  const float* pair = (const float*)d_in[1];
  auto P = [&](int i) { return (const float*)d_in[i]; };

  // ---- workspace arena (floats) ----
  float* ws = (float*)d_ws;
  long long off = 0;
  auto alloc = [&](long long nf) { float* p = ws + off; off += nf; return p; };
  float* M_CUR = alloc(SN * CM);       // mutable MSA
  float* Z_CUR = alloc(NN * CZ);       // mutable pair
  float* LNB   = alloc(SN * CM);       // layernorm output (ml / zl)
  float* BQ    = alloc(SN * HCM);
  float* BK    = alloc(SN * HCM);
  float* BV    = alloc(SN * HCM);
  float* BG    = alloc(SN * HCM);
  float* BO    = alloc(SN * HCM);
  float* BP    = alloc(SN * CM);       // projection outputs
  float* BIG   = alloc((long long)S * HM * NN);  // 67.1M: logits / hiddens / outer
  float* X1 = alloc(NN * CZ);
  float* X2 = alloc(NN * CZ);
  float* X3 = alloc(NN * CZ);
  float* X4 = alloc(NN * CZ);
  float* BB1 = alloc(NN * 8);
  float* BB2 = alloc(NN * 8);
  float* OA = alloc(SN * 32);
  float* OB = alloc(SN * 32);
  float* HS = alloc((long long)NR * CM);
  float* HT = alloc((long long)NR * CM);
  float* HY = alloc((long long)NR * CM);

  auto ln = [&](const float* x, int gi, int bi, float* y, long long rows, int C) {
    layernorm_k<<<(unsigned)rows, 128, 0, stream>>>(x, P(gi), P(bi), y, C);
  };
  auto gemm1 = [&](const float* A, const float* B, const float* bias, float* C,
                   int M, int N, int K, float alpha, int act) {
    gemm_launch(stream, A, B, bias, C, M, N, K, K, N, N,
                0, 0, 0, 0, 0, 0, 1, 1, alpha, act, 0, 0);
  };
  auto softmax = [&](float* x, const float* bias, int B2, int M, int ncols, long long rows) {
    softmax_k<<<(unsigned)rows, 128, 0, stream>>>(x, bias, B2, M, ncols);
  };
  auto ew = [&](long long n) { return dim3((unsigned)((n + 255) / 256)); };
  auto accum = [&](float* d, const float* s, long long n) {
    accum_k<<<ew(n), 256, 0, stream>>>(d, s, n);
  };
  auto mul = [&](float* o, const float* a, const float* b, long long n) {
    mul_k<<<ew(n), 256, 0, stream>>>(o, a, b, n);
  };
  auto transpose = [&](const float* in, float* out, int R, int C) {
    dim3 g((C + 31) / 32, (R + 31) / 32), b(32, 8);
    transpose_k<<<g, b, 0, stream>>>(in, out, R, C);
  };

  hipMemcpyAsync(M_CUR, msa,  SN * CM * sizeof(float), hipMemcpyDeviceToDevice, stream);
  hipMemcpyAsync(Z_CUR, pair, NN * CZ * sizeof(float), hipMemcpyDeviceToDevice, stream);

  // ===================== MSA row attention (pair-biased, gated) =====================
  ln(M_CUR, 2, 3, LNB, SN, CM);                 // ml
  ln(Z_CUR, 4, 5, BP, NN, CZ);                  // zl (BP is free here)
  gemm1(BP, P(13), nullptr, BB1, (int)NN, HM, CZ, 1.f, 0);    // [N*N, H]
  transpose(BB1, BB2, (int)NN, HM);                           // bias [H, N, N]
  gemm1(LNB, P(6), nullptr, BQ, (int)SN, HCM, CM, qscale, 0); // q
  gemm1(LNB, P(7), nullptr, BK, (int)SN, HCM, CM, 1.f, 0);    // k
  gemm1(LNB, P(8), nullptr, BV, (int)SN, HCM, CM, 1.f, 0);    // v
  gemm1(LNB, P(9), P(10),  BG, (int)SN, HCM, CM, 1.f, 2);     // g = sigmoid
  // scores[s,h,i,j] = q . k   (batch b1=s, b2=h)
  gemm_launch(stream, BQ, BK, nullptr, BIG, NR, NR, 32, HCM, HCM, NR,
              (long long)NR * HCM, 32, (long long)NR * HCM, 32,
              (long long)HM * NN, NN, S, HM, 1.f, 0, 0, 1);
  softmax(BIG, BB2, HM, NR, NR, (long long)S * HM * NR);
  // o[s,i,h,c] = a @ v
  gemm_launch(stream, BIG, BV, nullptr, BO, NR, 32, NR, NR, HCM, HCM,
              (long long)HM * NN, NN, (long long)NR * HCM, 32,
              (long long)NR * HCM, 32, S, HM, 1.f, 0, 0, 0);
  mul(BO, BG, BO, SN * HCM);
  gemm1(BO, P(11), P(12), BP, (int)SN, CM, HCM, 1.f, 0);
  accum(M_CUR, BP, SN * CM);

  // ===================== MSA column attention =====================
  ln(M_CUR, 14, 15, LNB, SN, CM);
  gemm1(LNB, P(16), nullptr, BQ, (int)SN, HCM, CM, qscale, 0);
  gemm1(LNB, P(17), nullptr, BK, (int)SN, HCM, CM, 1.f, 0);
  gemm1(LNB, P(18), nullptr, BV, (int)SN, HCM, CM, 1.f, 0);
  gemm1(LNB, P(19), P(20),  BG, (int)SN, HCM, CM, 1.f, 2);
  // scores[h,i,s,t]  (batch b1=h, b2=i)
  gemm_launch(stream, BQ, BK, nullptr, BIG, S, S, 32, NR * HCM, NR * HCM, S,
              32, HCM, 32, HCM,
              (long long)NR * S * S, (long long)S * S, HM, NR, 1.f, 0, 0, 1);
  softmax(BIG, nullptr, 1, 1, S, (long long)HM * NR * S);
  // o[s,i,h,c]
  gemm_launch(stream, BIG, BV, nullptr, BO, S, 32, S, S, NR * HCM, NR * HCM,
              (long long)NR * S * S, (long long)S * S, 32, HCM,
              32, HCM, HM, NR, 1.f, 0, 0, 0);
  mul(BO, BG, BO, SN * HCM);
  gemm1(BO, P(21), P(22), BP, (int)SN, CM, HCM, 1.f, 0);
  accum(M_CUR, BP, SN * CM);

  // ===================== MSA transition =====================
  ln(M_CUR, 23, 24, LNB, SN, CM);
  gemm1(LNB, P(25), P(26), BIG, (int)SN, NT * CM, CM, 1.f, 1);  // relu
  gemm1(BIG, P(27), P(28), BP, (int)SN, CM, NT * CM, 1.f, 0);
  accum(M_CUR, BP, SN * CM);

  // ===================== Outer product mean =====================
  ln(M_CUR, 29, 30, LNB, SN, CM);
  gemm1(LNB, P(31), P(32), OA, (int)SN, 32, CM, 1.f, 0);
  gemm1(LNB, P(33), P(34), OB, (int)SN, 32, CM, 1.f, 0);
  // outer[i,j,c,d] = sum_s a[s,i,c] b[s,j,d]   (TN, batch b1=i, b2=j)
  gemm_launch(stream, OA, OB, nullptr, BIG, 32, 32, S, NR * 32, NR * 32, 32,
              32, 0, 0, 32, (long long)NR * 1024, 1024, NR, NR, 1.f, 0, 1, 0);
  gemm1(BIG, P(35), P(36), BP, (int)NN, CZ, 1024, 1.f, 0);
  accum_scaled_k<<<ew(NN * CZ), 256, 0, stream>>>(Z_CUR, BP, 1.f / (S + 1e-3f), NN * CZ);

  // ===================== Triangle multiplicative updates =====================
  auto tri_mult = [&](int base, bool outgoing) {
    ln(Z_CUR, base + 0, base + 1, LNB, NN, CZ);                   // zl
    gemm1(LNB, P(base + 6), P(base + 7), BQ, (int)NN, CZ, CZ, 1.f, 2);  // sig(ag)
    gemm1(LNB, P(base + 4), P(base + 5), BK, (int)NN, CZ, CZ, 1.f, 0);  // ap
    mul(X1, BQ, BK, NN * CZ);                                     // a
    gemm1(LNB, P(base + 10), P(base + 11), BQ, (int)NN, CZ, CZ, 1.f, 2);// sig(bg)
    gemm1(LNB, P(base + 8),  P(base + 9),  BK, (int)NN, CZ, CZ, 1.f, 0);// bp
    mul(X2, BQ, BK, NN * CZ);                                     // b
    gemm1(LNB, P(base + 12), P(base + 13), BG, (int)NN, CZ, CZ, 1.f, 2);// g
    transpose(X1, X3, (int)NN, CZ);   // at[c, i, k]  (rows were (i,k) or (k,i))
    transpose(X2, X4, (int)NN, CZ);   // bt
    // per-channel 256x256x256 GEMM (batch b1=c)
    if (outgoing)  // out[i,j] = sum_k at[c][i][k] * bt[c][j][k]  (NT)
      gemm_launch(stream, X3, X4, nullptr, X1, NR, NR, NR, NR, NR, NR,
                  NN, 0, NN, 0, NN, 0, CZ, 1, 1.f, 0, 0, 1);
    else           // out[i,j] = sum_k at[c][k][i] * bt[c][k][j]  (TN)
      gemm_launch(stream, X3, X4, nullptr, X1, NR, NR, NR, NR, NR, NR,
                  NN, 0, NN, 0, NN, 0, CZ, 1, 1.f, 0, 1, 0);
    transpose(X1, X2, CZ, (int)NN);   // x[i,j,c]
    ln(X2, base + 2, base + 3, BV, NN, CZ);                       // layernorm lno
    gemm1(BV, P(base + 14), P(base + 15), BK, (int)NN, CZ, CZ, 1.f, 0);
    mul(BK, BG, BK, NN * CZ);
    accum(Z_CUR, BK, NN * CZ);
  };
  tri_mult(37, true);    // outgoing
  tri_mult(53, false);   // incoming

  // ===================== Triangle attention =====================
  auto tri_attn = [&](const float* zsrc, int pb, float* outbuf) {
    ln(zsrc, pb + 0, pb + 1, LNB, NN, CZ);
    gemm1(LNB, P(pb + 9), nullptr, BB1, (int)NN, HP, CZ, 1.f, 0);
    transpose(BB1, BB2, (int)NN, HP);                    // bias [H, N, N]
    gemm1(LNB, P(pb + 2), nullptr, BQ, (int)NN, HCP, CZ, qscale, 0);
    gemm1(LNB, P(pb + 3), nullptr, BK, (int)NN, HCP, CZ, 1.f, 0);
    gemm1(LNB, P(pb + 4), nullptr, BV, (int)NN, HCP, CZ, 1.f, 0);
    gemm1(LNB, P(pb + 5), P(pb + 6), BG, (int)NN, HCP, CZ, 1.f, 2);
    // scores[i,h,j,k]  (batch b1=i, b2=h)
    gemm_launch(stream, BQ, BK, nullptr, BIG, NR, NR, 32, HCP, HCP, NR,
                (long long)NR * HCP, 32, (long long)NR * HCP, 32,
                (long long)HP * NN, NN, NR, HP, 1.f, 0, 0, 1);
    softmax(BIG, BB2, HP, NR, NR, (long long)NR * HP * NR);
    // o[i,j,h,c]
    gemm_launch(stream, BIG, BV, nullptr, BO, NR, 32, NR, NR, HCP, HCP,
                (long long)HP * NN, NN, (long long)NR * HCP, 32,
                (long long)NR * HCP, 32, NR, HP, 1.f, 0, 0, 0);
    mul(BO, BG, BO, NN * HCP);
    gemm1(BO, P(pb + 7), P(pb + 8), outbuf, (int)NN, CZ, HCP, 1.f, 0);
  };
  // starting node
  tri_attn(Z_CUR, 69, BP);
  accum(Z_CUR, BP, NN * CZ);
  // ending node: operate on z^T, permute result back
  permute01_k<<<ew(NN * CZ), 256, 0, stream>>>(Z_CUR, X1, NR, NR, CZ);
  tri_attn(X1, 79, BP);
  permute01_k<<<ew(NN * CZ), 256, 0, stream>>>(BP, X2, NR, NR, CZ);
  accum(Z_CUR, X2, NN * CZ);

  // ===================== Pair transition =====================
  ln(Z_CUR, 89, 90, LNB, NN, CZ);
  gemm1(LNB, P(91), P(92), BIG, (int)NN, NT * CZ, CZ, 1.f, 1);
  gemm1(BIG, P(93), P(94), BP, (int)NN, CZ, NT * CZ, 1.f, 0);
  accum(Z_CUR, BP, NN * CZ);

  // ===================== Single-representation head =====================
  gemm1(M_CUR, P(95), P(96), HS, NR, CM, CM, 1.f, 0);    // s = m[0] @ ws + bs
  gemm1(HS, P(97), P(98), HT, NR, CM, CM, 1.f, 1);       // relu(s@l1+b1)
  add2_k<<<ew((long long)NR * CM), 256, 0, stream>>>(HY, HS, HT, (long long)NR * CM);
  gemm1(HY, P(99), P(100), (float*)d_out, NR, 20, CM, 1.f, 0);
  softmax((float*)d_out, nullptr, 1, 1, 20, NR);
}
